// MultiHeadAttention_4234837753857
// MI455X (gfx1250) — matrix operand order repaired and checked by reading, not compiled, not run on hardware
//
#include <hip/hip_runtime.h>

typedef __attribute__((ext_vector_type(16))) __bf16 bf16x16;
typedef __attribute__((ext_vector_type(8)))  __bf16 bf16x8;
typedef __attribute__((ext_vector_type(8)))  float  f32x8;

#define B_   2
#define N_   2048
#define C_   1024
#define H_   16
#define HD_  64
#define SCALE_ 0.125f   // HD^-0.5

__device__ __forceinline__ __bf16 f2bf(float f) {
  unsigned u = __builtin_bit_cast(unsigned, f);
  u += 0x7FFFu + ((u >> 16) & 1u);                 // round-to-nearest-even
  unsigned short s = (unsigned short)(u >> 16);
  return __builtin_bit_cast(__bf16, s);
}

__device__ __forceinline__ bf16x16 cat8(bf16x8 lo, bf16x8 hi) {
  return __builtin_shufflevector(lo, hi, 0,1,2,3,4,5,6,7,8,9,10,11,12,13,14,15);
}

__device__ __forceinline__ f32x8 wmma_bf16(bf16x16 a, bf16x16 b, f32x8 c) {
  // D = A(16x32 bf16) * B(32x16 bf16) + C(16x16 f32)
  return __builtin_amdgcn_wmma_f32_16x16x32_bf16(false, a, false, b, (short)0, c,
                                                 false, false);
}

// ---------------------------------------------------------------- convert
__global__ void cvt_f32_bf16(const float* __restrict__ in, __bf16* __restrict__ out,
                             int n) {
  int i = blockIdx.x * blockDim.x + threadIdx.x;
  if (i < n) out[i] = f2bf(in[i]);
}

// ------------------------------------------------- 32x64 GEMM building blocks
// a0/a1 already point at (row base + hw*8); bcol at (col base + hw*8).
__device__ __forceinline__ void load_a2(const __bf16* a0, const __bf16* a1,
                                        int kk, bf16x16 A[2]) {
  A[0] = cat8(*(const bf16x8*)(a0 + kk), *(const bf16x8*)(a0 + kk + 16));
  A[1] = cat8(*(const bf16x8*)(a1 + kk), *(const bf16x8*)(a1 + kk + 16));
}
__device__ __forceinline__ void load_b4(const __bf16* bcol, int kk, bf16x16 B[4]) {
#pragma unroll
  for (int c = 0; c < 4; ++c)
    B[c] = cat8(*(const bf16x8*)(bcol + (size_t)c * 16 * C_ + kk),
                *(const bf16x8*)(bcol + (size_t)c * 16 * C_ + kk + 16));
}
__device__ __forceinline__ void mm8(f32x8 acc[2][4], const bf16x16 A[2],
                                    const bf16x16 B[4]) {
#pragma unroll
  for (int c = 0; c < 4; ++c) {
    acc[0][c] = wmma_bf16(A[0], B[c], acc[0][c]);
    acc[1][c] = wmma_bf16(A[1], B[c], acc[1][c]);
  }
}

// Double-buffered K loop: loads for block k+32 overlap WMMAs on block k.
__device__ __forceinline__ void gemm_loop(const __bf16* a0, const __bf16* a1,
                                          const __bf16* bcol, f32x8 acc[2][4]) {
  bf16x16 AX[2], BX[4], AY[2], BY[4];
  load_a2(a0, a1, 0, AX);
  load_b4(bcol, 0, BX);
#pragma unroll 1
  for (int kk = 0; kk < C_; kk += 64) {
    load_a2(a0, a1, kk + 32, AY);
    load_b4(bcol, kk + 32, BY);
    mm8(acc, AX, BX);
    if (kk + 64 < C_) {
      load_a2(a0, a1, kk + 64, AX);
      load_b4(bcol, kk + 64, BX);
    }
    mm8(acc, AY, BY);
  }
}

// ---------------------------------------------------------------- QKV GEMM
// qkv[m,d] = sum_c x[m,c] * w_qkv[d,c]   (M=4096, K=1024, Nd=3072)
// Wave computes a 32x64 tile; routes output into q (pre-scaled), k, vT.
__global__ __launch_bounds__(256) void qkv_gemm(const __bf16* __restrict__ xb,
                                                const __bf16* __restrict__ wq,
                                                __bf16* __restrict__ qo,
                                                __bf16* __restrict__ ko,
                                                __bf16* __restrict__ vto) {
  int wave = (blockIdx.x * blockDim.x + threadIdx.x) >> 5;
  int lane = threadIdx.x & 31;
  int hw = lane >> 4, lm = lane & 15;
  int nt = wave % 48, mt = wave / 48;      // 48 n-tiles, 128 m-tiles
  int m0 = mt * 32, d0 = nt * 64;

  const __bf16* a0   = xb + (size_t)(m0 + lm) * C_ + hw * 8;
  const __bf16* a1   = xb + (size_t)(m0 + 16 + lm) * C_ + hw * 8;
  const __bf16* bcol = wq + (size_t)(d0 + lm) * C_ + hw * 8;

  f32x8 acc[2][4] = {};
  gemm_loop(a0, a1, bcol, acc);

#pragma unroll
  for (int r = 0; r < 2; ++r)
#pragma unroll
    for (int c = 0; c < 4; ++c) {
      int d = d0 + c * 16 + lm;
      int which = d >> 10, dd = d & 1023, h = dd >> 6, hd = dd & 63;
#pragma unroll
      for (int j = 0; j < 8; ++j) {
        int mr = m0 + r * 16 + j + hw * 8;
        int b = mr >> 11, n = mr & 2047;
        float v = acc[r][c][j];
        size_t bhh = (size_t)(b * H_ + h);
        if (which == 0)      qo[(bhh * N_ + n) * HD_ + hd] = f2bf(v * SCALE_);
        else if (which == 1) ko[(bhh * N_ + n) * HD_ + hd] = f2bf(v);
        else                 vto[(bhh * HD_ + hd) * N_ + n] = f2bf(v);
      }
    }
}

// ---------------------------------------------------------------- attention
// Flash-style: wave owns 16 queries, streams 32 keys/iter with online softmax.
__global__ __launch_bounds__(256) void attn_kernel(const __bf16* __restrict__ q,
                                                   const __bf16* __restrict__ k,
                                                   const __bf16* __restrict__ vt,
                                                   __bf16* __restrict__ ao) {
  __shared__ __bf16 ldsP[8 * 16 * 32];     // per-wave 16x32 P tile
  const int w = threadIdx.x >> 5;
  const int lane = threadIdx.x & 31;
  const int hw = lane >> 4, lm = lane & 15;
  const int qblk = blockIdx.x & 15;
  const int bh = blockIdx.x >> 4;          // b*16+h
  const int b = bh >> 4, h = bh & 15;
  const int q0 = qblk * 128 + w * 16;

  const __bf16* qp = q  + ((size_t)bh * N_ + q0) * HD_;
  const __bf16* kp = k  + (size_t)bh * N_ * HD_;
  const __bf16* vp = vt + (size_t)bh * HD_ * N_;
  __bf16* op = ao + ((size_t)(b * N_ + q0)) * C_ + h * HD_;
  __bf16* lds = ldsP + w * (16 * 32);

  // Q as two A-fragments (K = d in [0,32) and [32,64))
  const __bf16* qrow = qp + (size_t)lm * HD_ + hw * 8;
  bf16x16 qa0 = cat8(*(const bf16x8*)(qrow),      *(const bf16x8*)(qrow + 16));
  bf16x16 qa1 = cat8(*(const bf16x8*)(qrow + 32), *(const bf16x8*)(qrow + 48));

  f32x8 o[4] = {};
  float mrow[8], lrow[8];
#pragma unroll
  for (int j = 0; j < 8; ++j) { mrow[j] = -1e30f; lrow[j] = 0.f; }

  auto load_k4 = [&](int kb, bf16x16 KB[4]) {
    const __bf16* k0 = kp + (size_t)(kb + lm) * HD_ + hw * 8;
    const __bf16* k1 = kp + (size_t)(kb + 16 + lm) * HD_ + hw * 8;
    KB[0] = cat8(*(const bf16x8*)(k0),      *(const bf16x8*)(k0 + 16));
    KB[1] = cat8(*(const bf16x8*)(k0 + 32), *(const bf16x8*)(k0 + 48));
    KB[2] = cat8(*(const bf16x8*)(k1),      *(const bf16x8*)(k1 + 16));
    KB[3] = cat8(*(const bf16x8*)(k1 + 32), *(const bf16x8*)(k1 + 48));
  };

  bf16x16 KB[4];
  load_k4(0, KB);

#pragma unroll 1
  for (int kb = 0; kb < N_; kb += 32) {
    // V-fragments issued now, consumed after the LDS round trip (overlap softmax)
    bf16x16 VB[4];
#pragma unroll
    for (int c = 0; c < 4; ++c)
      VB[c] = cat8(*(const bf16x8*)(vp + (size_t)(c * 16 + lm) * N_ + kb + hw * 8),
                   *(const bf16x8*)(vp + (size_t)(c * 16 + lm) * N_ + kb + hw * 8 + 16));

    f32x8 s0 = {}, s1 = {};
    s0 = wmma_bf16(qa0, KB[0], s0);
    s0 = wmma_bf16(qa1, KB[1], s0);
    s1 = wmma_bf16(qa0, KB[2], s1);
    s1 = wmma_bf16(qa1, KB[3], s1);

    // prefetch next key block during softmax
    if (kb + 32 < N_) load_k4(kb + 32, KB);

    float alpha[8];
#pragma unroll
    for (int j = 0; j < 8; ++j) {
      float cur = fmaxf(s0[j], s1[j]);
      cur = fmaxf(cur, __shfl_xor(cur, 1));
      cur = fmaxf(cur, __shfl_xor(cur, 2));
      cur = fmaxf(cur, __shfl_xor(cur, 4));
      cur = fmaxf(cur, __shfl_xor(cur, 8));
      float mn = fmaxf(mrow[j], cur);
      float a  = __expf(mrow[j] - mn);
      float p0 = __expf(s0[j] - mn);
      float p1 = __expf(s1[j] - mn);
      float rs = p0 + p1;
      rs += __shfl_xor(rs, 1);
      rs += __shfl_xor(rs, 2);
      rs += __shfl_xor(rs, 4);
      rs += __shfl_xor(rs, 8);
      lrow[j] = lrow[j] * a + rs;
      mrow[j] = mn;
      alpha[j] = a;
      s0[j] = p0; s1[j] = p1;
    }
#pragma unroll
    for (int c = 0; c < 4; ++c)
#pragma unroll
      for (int j = 0; j < 8; ++j) o[c][j] *= alpha[j];

    // C-layout -> A-layout via per-wave LDS tile
#pragma unroll
    for (int j = 0; j < 8; ++j) {
      int m = j + hw * 8;
      lds[m * 32 + lm]      = f2bf(s0[j]);
      lds[m * 32 + 16 + lm] = f2bf(s1[j]);
    }
    asm volatile("s_wait_dscnt 0x0" ::: "memory");   // DS in-order per wave
    const __bf16* prow = lds + lm * 32 + hw * 8;
    bf16x16 pa = cat8(*(const bf16x8*)(prow), *(const bf16x8*)(prow + 16));
#pragma unroll
    for (int c = 0; c < 4; ++c)
      o[c] = wmma_bf16(pa, VB[c], o[c]);
  }
#pragma unroll
  for (int c = 0; c < 4; ++c)
#pragma unroll
    for (int j = 0; j < 8; ++j) {
      int m = j + hw * 8;
      op[(size_t)m * C_ + c * 16 + lm] = f2bf(o[c][j] / lrow[j]);
    }
}

// ---------------------------------------------------------------- proj GEMM
__global__ __launch_bounds__(256) void proj_gemm(const __bf16* __restrict__ ab,
                                                 const __bf16* __restrict__ wp,
                                                 const float* __restrict__ bias,
                                                 float* __restrict__ out) {
  int wave = (blockIdx.x * blockDim.x + threadIdx.x) >> 5;
  int lane = threadIdx.x & 31;
  int hw = lane >> 4, lm = lane & 15;
  int nt = wave & 15, mt = wave >> 4;      // 16 n-tiles, 128 m-tiles
  int m0 = mt * 32, d0 = nt * 64;

  const __bf16* a0   = ab + (size_t)(m0 + lm) * C_ + hw * 8;
  const __bf16* a1   = ab + (size_t)(m0 + 16 + lm) * C_ + hw * 8;
  const __bf16* bcol = wp + (size_t)(d0 + lm) * C_ + hw * 8;

  f32x8 acc[2][4] = {};
  gemm_loop(a0, a1, bcol, acc);

#pragma unroll
  for (int r = 0; r < 2; ++r)
#pragma unroll
    for (int c = 0; c < 4; ++c) {
      int d = d0 + c * 16 + lm;
      float bb = bias[d];
#pragma unroll
      for (int j = 0; j < 8; ++j) {
        int m = m0 + r * 16 + j + hw * 8;
        out[(size_t)m * C_ + d] = acc[r][c][j] + bb;
      }
    }
}

// ---------------------------------------------------------------- launch
extern "C" void kernel_launch(void* const* d_in, const int* in_sizes, int n_in,
                              void* d_out, int out_size, void* d_ws, size_t ws_size,
                              hipStream_t stream) {
  (void)in_sizes; (void)n_in; (void)out_size; (void)ws_size;
  const float* x      = (const float*)d_in[0];
  const float* w_qkv  = (const float*)d_in[1];
  const float* w_proj = (const float*)d_in[2];
  const float* b_proj = (const float*)d_in[3];
  float* out = (float*)d_out;

  char* ws = (char*)d_ws;
  size_t off = 0;
  auto take = [&](size_t nelem) {
    char* p = ws + off;
    off = (off + nelem * 2 + 255) & ~(size_t)255;
    return (__bf16*)p;
  };
  const size_t BN = (size_t)B_ * N_;                 // 4096 rows
  __bf16* xb  = take(BN * C_);                       // x bf16
  __bf16* wqb = take((size_t)3 * C_ * C_);           // w_qkv bf16
  __bf16* wpb = take((size_t)C_ * C_);               // w_proj bf16
  __bf16* qb  = take(BN * C_);                       // q  [b,h,n,hd] (pre-scaled)
  __bf16* kb  = take(BN * C_);                       // k  [b,h,n,hd]
  __bf16* vtb = take(BN * C_);                       // vT [b,h,hd,n]
  __bf16* aob = take(BN * C_);                       // attn out [b,n,h*hd]

  int n1 = (int)(BN * C_);           // 4,194,304
  int n2 = 3 * C_ * C_;              // 3,145,728
  int n3 = C_ * C_;                  // 1,048,576
  cvt_f32_bf16<<<(n1 + 255) / 256, 256, 0, stream>>>(x, xb, n1);
  cvt_f32_bf16<<<(n2 + 255) / 256, 256, 0, stream>>>(w_qkv, wqb, n2);
  cvt_f32_bf16<<<(n3 + 255) / 256, 256, 0, stream>>>(w_proj, wpb, n3);

  // 128 m-tiles * 48 n-tiles = 6144 waves / 8 per block
  qkv_gemm<<<768, 256, 0, stream>>>(xb, wqb, qb, kb, vtb);
  // B*H*(N/128) = 512 blocks, 8 waves * 16 queries each
  attn_kernel<<<512, 256, 0, stream>>>(qb, kb, vtb, aob);
  // 128 m-tiles * 16 n-tiles = 2048 waves / 8 per block
  proj_gemm<<<256, 256, 0, stream>>>(aob, wpb, b_proj, out);
}
